// GraphTransConv_30468497997918
// MI455X (gfx1250) — matrix operand order repaired and checked by reading, not compiled, not run on hardware
//
#include <hip/hip_runtime.h>
#include <hip/hip_bf16.h>

typedef __attribute__((ext_vector_type(16))) _Float16 v16h;
typedef __attribute__((ext_vector_type(8)))  float    v8f;

#define NNODES 8192
#define CCH    256
#define NHEADS 4
#define KDEG   16

union FragAB   { v16h v; uint32_t u[8]; };
union FragC    { v8f  v; float    f[8]; };
union HalfPair { uint32_t u; _Float16 h[2]; };

// ---------------------------------------------------------------- converts
__global__ void cvt_f32_to_f16(const float* __restrict__ in,
                               _Float16* __restrict__ out, int n) {
  int i = blockIdx.x * blockDim.x + threadIdx.x;
  int stride = gridDim.x * blockDim.x;
  for (; i < n; i += stride) out[i] = (_Float16)in[i];
}

// ---------------------------------------------------------------- WMMA GEMM
// OUT[M,Nc] = A[M,256] @ W[256,Nc], f16 inputs, f32 accumulate.
// Block = 256 threads = 8 wave32s; block tile 64(M) x 128(N).
// Wave w: m-strip = (w>>1)*16, n-half = (w&1)*64; computes 4 16x16 tiles,
// reusing one A fragment across 4 B fragments (2.5 b128 DS loads per WMMA),
// 4 independent accumulators to hide WMMA latency. 32 WMMAs per wave.
__global__ __launch_bounds__(256)
void gemm_wmma(const _Float16* __restrict__ A, const _Float16* __restrict__ W,
               const float* __restrict__ bias, float* __restrict__ outF,
               _Float16* __restrict__ outH, int Nc, int relu) {
  __shared__ uint32_t sA[64 * 128];    // [m][kpair]   64 rows x 256 halves (32KB)
  __shared__ uint32_t sW[128 * 128];   // [n][kpair]  128 cols x 256 halves (64KB, W^T)
  const int tid = threadIdx.x;
  const int m0 = blockIdx.x * 64;
  const int n0 = blockIdx.y * 128;
  const uint32_t* Au = (const uint32_t*)A;
  const uint32_t* Wu = (const uint32_t*)W;

  // A tile: rows m0..m0+63, all K (256 halves = 128 dwords per row)
  for (int i = 0; i < 32; ++i) {
    int idx = tid + i * 256;                  // 0..8191
    int m = idx >> 7, kp = idx & 127;
    sA[idx] = Au[(size_t)(m0 + m) * 128 + kp];
  }
  // W tile, stored transposed: sW halves laid out [n][k], k contiguous
  const int NcU = Nc >> 1;
  _Float16* sWh = (_Float16*)sW;
  for (int i = 0; i < 64; ++i) {
    int idx = tid + i * 256;                  // 0..16383
    int k = idx >> 6, np = idx & 63;          // 64 dwords (=128 n) per k row
    HalfPair p; p.u = Wu[(size_t)k * NcU + (n0 >> 1) + np];
    int n = np * 2;
    sWh[n * 256 + k]       = p.h[0];
    sWh[(n + 1) * 256 + k] = p.h[1];
  }
  __syncthreads();

  const int wave = tid >> 5, lane = tid & 31;
  const int msub = wave >> 1;                 // 0..3 -> m-strip
  const int nhalf = (wave & 1) * 64;          // 0 or 64 -> n-half
  const int grp = lane >> 4, row = lane & 15;

  FragC acc[4];
#pragma unroll
  for (int t = 0; t < 4; ++t)
#pragma unroll
    for (int r = 0; r < 8; ++r) acc[t].f[r] = 0.0f;

#pragma unroll
  for (int kk = 0; kk < 8; ++kk) {
    const int KB = kk * 32;
    FragAB a;
#pragma unroll
    for (int v = 0; v < 8; ++v) {
      // A fragment, contiguous k order used here (same as B): VGPR v -> K=2v+16*grp
      int ka = 2 * v + 16 * grp + KB;
      a.u[v] = sA[(msub * 16 + row) * 128 + (ka >> 1)];
    }
#pragma unroll
    for (int t = 0; t < 4; ++t) {
      FragAB b;
#pragma unroll
      for (int v = 0; v < 8; ++v) {
        // B fragment, contiguous k order used here (same as A): VGPR v -> K=2v+16*grp, column = lane&15
        int kb = 2 * v + 16 * grp + KB;
        b.u[v] = sW[(nhalf + t * 16 + row) * 128 + (kb >> 1)];
      }
      acc[t].v = __builtin_amdgcn_wmma_f32_16x16x32_f16(
          false, a.v, false, b.v, (short)0, acc[t].v, false, false);
    }
  }

  // C/D layout: VGPR r -> M = r + 8*grp, N = lane&15
#pragma unroll
  for (int t = 0; t < 4; ++t) {
#pragma unroll
    for (int r = 0; r < 8; ++r) {
      int gm = m0 + msub * 16 + r + 8 * grp;
      int gn = n0 + nhalf + t * 16 + row;
      float val = acc[t].f[r];
      if (bias) val += bias[gn];
      if (relu) val = fmaxf(val, 0.0f);
      size_t o = (size_t)gm * Nc + gn;
      if (outF) outF[o] = val;
      if (outH) outH[o] = (_Float16)val;
    }
  }
}

// ---------------------------------------------------------------- graph prop
// H[i,:] = relu( (G[i,:] + sum_{e in-edges of i} G[src_e,:]) / 17 ); A16 = f16(H)
__global__ __launch_bounds__(256)
void prop_relu(const float* __restrict__ G, const int* __restrict__ srcIdx,
               float* __restrict__ H, _Float16* __restrict__ A16) {
  __shared__ int sSrc[KDEG];
  const int i = blockIdx.x, c = threadIdx.x;
  if (c < KDEG) sSrc[c] = srcIdx[i * KDEG + c];
  __syncthreads();
  float acc = G[(size_t)i * CCH + c];
#pragma unroll
  for (int e = 0; e < KDEG; ++e) acc += G[(size_t)sSrc[e] * CCH + c];
  acc *= (1.0f / (1.0f + (float)KDEG));
  acc = fmaxf(acc, 0.0f);
  H[(size_t)i * CCH + c] = acc;
  A16[(size_t)i * CCH + c] = (_Float16)acc;
}

// ---------------------------------------------------------------- attention
// One block (128 thr = 4 wave32) per node; wave w = head w. 16-edge softmax.
__global__ __launch_bounds__(128)
void attn_kernel(const _Float16* __restrict__ Q, const _Float16* __restrict__ Km,
                 const _Float16* __restrict__ Vm, const int* __restrict__ srcIdx,
                 float* __restrict__ ATT) {
  __shared__ int sSrc[KDEG];
  __shared__ float sAcc[NHEADS * CCH];
  const int i = blockIdx.x;
  const int wave = threadIdx.x >> 5, lane = threadIdx.x & 31;
  if (threadIdx.x < KDEG) sSrc[threadIdx.x] = srcIdx[i * KDEG + threadIdx.x];
  __syncthreads();

  const size_t qoff = (size_t)i * (NHEADS * CCH) + wave * CCH;
  float q[8];
#pragma unroll
  for (int j = 0; j < 8; ++j) q[j] = (float)Q[qoff + lane + 32 * j];

  float alpha[KDEG];
#pragma unroll
  for (int e = 0; e < KDEG; ++e) {
    const size_t koff = (size_t)sSrc[e] * (NHEADS * CCH) + wave * CCH;
    float p = 0.0f;
#pragma unroll
    for (int j = 0; j < 8; ++j) p += q[j] * (float)Km[koff + lane + 32 * j];
#pragma unroll
    for (int off = 16; off >= 1; off >>= 1) p += __shfl_xor(p, off, 32);
    alpha[e] = p * 0.0625f;   // 1/sqrt(256)
  }
  float m = alpha[0];
#pragma unroll
  for (int e = 1; e < KDEG; ++e) m = fmaxf(m, alpha[e]);
  float s = 0.0f;
#pragma unroll
  for (int e = 0; e < KDEG; ++e) { alpha[e] = __expf(alpha[e] - m); s += alpha[e]; }
  const float inv_s = 1.0f / s;

  float acc[8];
#pragma unroll
  for (int j = 0; j < 8; ++j) acc[j] = 0.0f;
#pragma unroll
  for (int e = 0; e < KDEG; ++e) {
    const float a = alpha[e] * inv_s;
    const size_t voff = (size_t)sSrc[e] * (NHEADS * CCH) + wave * CCH;
#pragma unroll
    for (int j = 0; j < 8; ++j) acc[j] += a * (float)Vm[voff + lane + 32 * j];
  }
#pragma unroll
  for (int j = 0; j < 8; ++j) sAcc[wave * CCH + lane + 32 * j] = acc[j];
  __syncthreads();

  // head mean: 128 threads, 2 channels each
  for (int c = threadIdx.x; c < CCH; c += 128) {
    float o = 0.25f * (sAcc[c] + sAcc[CCH + c] + sAcc[2 * CCH + c] + sAcc[3 * CCH + c]);
    ATT[(size_t)i * CCH + c] = o;
  }
}

// ---------------------------------------------------------------- beta + LN
__device__ __forceinline__ float blockSum256(float v, float* red) {
  const int t = threadIdx.x;
  red[t] = v; __syncthreads();
#pragma unroll
  for (int s = 128; s > 0; s >>= 1) {
    if (t < s) red[t] += red[t + s];
    __syncthreads();
  }
  float r = red[0]; __syncthreads();
  return r;
}

__global__ __launch_bounds__(256)
void combine_ln(const float* __restrict__ ATT, const float* __restrict__ S,
                const float* __restrict__ Wb, const float* __restrict__ lng,
                const float* __restrict__ lnb, float* __restrict__ dst,
                _Float16* __restrict__ A16) {
  __shared__ float red[256];
  const int i = blockIdx.x, c = threadIdx.x;
  const float o  = ATT[(size_t)i * CCH + c];
  const float xr = S[(size_t)i * CCH + c];
  // beta = sigmoid([out, x_r, out-x_r] . Wbeta)
  float p = o * Wb[c] + xr * Wb[CCH + c] + (o - xr) * Wb[2 * CCH + c];
  const float tot = blockSum256(p, red);
  const float beta = 1.0f / (1.0f + __expf(-tot));
  const float y = beta * xr + (1.0f - beta) * o;
  // LayerNorm over C=256 then ReLU
  const float mu = blockSum256(y, red) * (1.0f / CCH);
  const float xc = y - mu;
  const float var = blockSum256(xc * xc, red) * (1.0f / CCH);
  float val = lng[c] * xc * rsqrtf(var + 1e-5f) + lnb[c];
  val = fmaxf(val, 0.0f);
  dst[(size_t)i * CCH + c] = val;
  A16[(size_t)i * CCH + c] = (_Float16)val;
}

// ---------------------------------------------------------------- launch
extern "C" void kernel_launch(void* const* d_in, const int* in_sizes, int n_in,
                              void* d_out, int out_size, void* d_ws, size_t ws_size,
                              hipStream_t stream) {
  (void)in_sizes; (void)n_in; (void)out_size; (void)ws_size;
  const float* x      = (const float*)d_in[0];
  const int*   srcIdx = (const int*)d_in[2];          // edge_index row 0 = src
  const float* gcW[2]   = { (const float*)d_in[3], (const float*)d_in[4] };
  const float* Wq[2]    = { (const float*)d_in[5],  (const float*)d_in[14] };
  const float* bq[2]    = { (const float*)d_in[6],  (const float*)d_in[15] };
  const float* Wk[2]    = { (const float*)d_in[7],  (const float*)d_in[16] };
  const float* bk[2]    = { (const float*)d_in[8],  (const float*)d_in[17] };
  const float* Wv[2]    = { (const float*)d_in[9],  (const float*)d_in[18] };
  const float* bv[2]    = { (const float*)d_in[10], (const float*)d_in[19] };
  const float* Ws[2]    = { (const float*)d_in[11], (const float*)d_in[20] };
  const float* bs[2]    = { (const float*)d_in[12], (const float*)d_in[21] };
  const float* Wbeta[2] = { (const float*)d_in[13], (const float*)d_in[22] };
  const float* lng[2]   = { (const float*)d_in[23], (const float*)d_in[25] };
  const float* lnb[2]   = { (const float*)d_in[24], (const float*)d_in[26] };

  // workspace carve
  char* base = (char*)d_ws;
  size_t off = 0;
  auto carve = [&](size_t bytes) { char* p = base + off;
                                   off += (bytes + 255) & ~(size_t)255; return p; };
  const size_t NC  = (size_t)NNODES * CCH;
  const size_t NHC = (size_t)NNODES * NHEADS * CCH;
  _Float16* A16 = (_Float16*)carve(NC * 2);
  _Float16* Q16 = (_Float16*)carve(NHC * 2);
  _Float16* K16 = (_Float16*)carve(NHC * 2);
  _Float16* V16 = (_Float16*)carve(NHC * 2);
  float*    G   = (float*)carve(NC * 4);
  float*    Sb  = (float*)carve(NC * 4);
  float*    ATT = (float*)carve(NC * 4);
  float*    H   = (float*)carve(NC * 4);
  _Float16* wGc[2], *wQ[2], *wK[2], *wV[2], *wS[2];
  for (int l = 0; l < 2; ++l) {
    wGc[l] = (_Float16*)carve(CCH * CCH * 2);
    wQ[l]  = (_Float16*)carve(CCH * NHEADS * CCH * 2);
    wK[l]  = (_Float16*)carve(CCH * NHEADS * CCH * 2);
    wV[l]  = (_Float16*)carve(CCH * NHEADS * CCH * 2);
    wS[l]  = (_Float16*)carve(CCH * CCH * 2);
  }

  // weight conversions (f32 -> f16)
  const int CT = 256;
  auto cvt = [&](const float* src, _Float16* dstp, int n) {
    cvt_f32_to_f16<<<(n + CT * 8 - 1) / (CT * 8), CT, 0, stream>>>(src, dstp, n);
  };
  for (int l = 0; l < 2; ++l) {
    cvt(gcW[l], wGc[l], CCH * CCH);
    cvt(Wq[l],  wQ[l],  CCH * NHEADS * CCH);
    cvt(Wk[l],  wK[l],  CCH * NHEADS * CCH);
    cvt(Wv[l],  wV[l],  CCH * NHEADS * CCH);
    cvt(Ws[l],  wS[l],  CCH * CCH);
  }
  cvt(x, A16, (int)NC);   // layer-1 activation in f16

  const dim3 blk(256);
  const dim3 gC(NNODES / 64, CCH / 128);                 // Nc = 256 GEMMs
  const dim3 gH(NNODES / 64, (NHEADS * CCH) / 128);      // Nc = 1024 GEMMs

  for (int l = 0; l < 2; ++l) {
    // GraphConv: G = A16 @ gcW ; H = relu(P @ G); A16 = f16(H)
    gemm_wmma<<<gC, blk, 0, stream>>>(A16, wGc[l], nullptr, G, nullptr, CCH, 0);
    prop_relu<<<NNODES, 256, 0, stream>>>(G, srcIdx, H, A16);
    // TransformerConv projections
    gemm_wmma<<<gH, blk, 0, stream>>>(A16, wQ[l], bq[l], nullptr, Q16, NHEADS * CCH, 0);
    gemm_wmma<<<gH, blk, 0, stream>>>(A16, wK[l], bk[l], nullptr, K16, NHEADS * CCH, 0);
    gemm_wmma<<<gH, blk, 0, stream>>>(A16, wV[l], bv[l], nullptr, V16, NHEADS * CCH, 0);
    gemm_wmma<<<gC, blk, 0, stream>>>(A16, wS[l], bs[l], Sb, nullptr, CCH, 0);
    // edge softmax attention + head mean
    attn_kernel<<<NNODES, 128, 0, stream>>>(Q16, K16, V16, srcIdx, ATT);
    // beta gate + LayerNorm + ReLU  (final layer writes d_out)
    float* dst = (l == 1) ? (float*)d_out : H;
    combine_ln<<<NNODES, 256, 0, stream>>>(ATT, Sb, Wbeta[l], lng[l], lnb[l], dst, A16);
  }
}
